// SimpleGraphCenteredNet_73375221284883
// MI455X (gfx1250) — compile-verified, matrix operand order checked by reading, not run on hardware
//
#include <hip/hip_runtime.h>
#include <hip/hip_bf16.h>

typedef __attribute__((ext_vector_type(16))) _Float16 v16h;
typedef __attribute__((ext_vector_type(8)))  _Float16 v8h;
typedef __attribute__((ext_vector_type(8)))  float    v8f;

#define HID 64

// ---------- degree / normalization ----------
__global__ __launch_bounds__(256) void k_deg_init(float* deg, int n) {
    int i = blockIdx.x * 256 + threadIdx.x;
    if (i < n) deg[i] = 1.0f;                       // self-loop contributes 1
}
__global__ __launch_bounds__(256) void k_deg_count(float* deg, const int* __restrict__ dst, int e) {
    int i = blockIdx.x * 256 + threadIdx.x;
    if (i < e) atomicAdd(&deg[dst[i]], 1.0f);
}
__global__ __launch_bounds__(256) void k_rsqrt_inplace(float* deg, int n) {
    int i = blockIdx.x * 256 + threadIdx.x;
    if (i < n) deg[i] = rsqrtf(deg[i]);             // deg >= 1 always (self-loops)
}

// ---------- layer 0: t = x[N,16] @ Wi[16,64] ----------
__global__ __launch_bounds__(256) void k_gemm_in(const float* __restrict__ X,
                                                 const float* __restrict__ W,
                                                 float* __restrict__ T, int n) {
    int idx = blockIdx.x * 256 + threadIdx.x;       // node*64 + c
    if (idx >= n * HID) return;
    int i = idx >> 6, c = idx & 63;
    const float* xr = X + (size_t)i * 16;
    float s = 0.f;
#pragma unroll
    for (int k = 0; k < 16; ++k) s = fmaf(xr[k], W[k * HID + c], s);
    T[idx] = s;
}

// ---------- WMMA GEMM: T[N,64] = H[N,64] @ W[64,64]  (f16 operands, f32 acc) ----------
__global__ __launch_bounds__(256) void k_gemm_wmma(const float* __restrict__ H,
                                                   const float* __restrict__ W,
                                                   float* __restrict__ T, int n) {
    // stage W transposed into LDS as f16: Wt[col*64 + k] = W[k*64 + col]
    __shared__ __align__(16) _Float16 Wt[HID * HID];
    for (int idx = threadIdx.x; idx < HID * HID; idx += 256) {
        int k = idx >> 6, c = idx & 63;
        Wt[c * HID + k] = (_Float16)W[idx];
    }
    __syncthreads();

    int wave = threadIdx.x >> 5;
    int lane = threadIdx.x & 31;
    int row0 = (blockIdx.x * 8 + wave) * 16;        // 16 rows per wave
    if (row0 >= n) return;                          // wave-uniform -> EXEC stays all-1s
    int m  = lane & 15;
    int hs = lane >> 4;

    const float* hrow = H + (size_t)(row0 + m) * HID;
    v8f cacc[4] = {};

#pragma unroll
    for (int kc = 0; kc < 2; ++kc) {                // two K=32 chunks
        int kb = kc * 32 + hs * 8;
        // A fragment: lane m holds K = kb..kb+7 (halves 0-7) and kb+16..kb+23 (halves 8-15)
        float4 a0 = *(const float4*)(hrow + kb);
        float4 a1 = *(const float4*)(hrow + kb + 4);
        float4 a2 = *(const float4*)(hrow + kb + 16);
        float4 a3 = *(const float4*)(hrow + kb + 20);
        v16h a;
        a[0]=(_Float16)a0.x;  a[1]=(_Float16)a0.y;  a[2]=(_Float16)a0.z;  a[3]=(_Float16)a0.w;
        a[4]=(_Float16)a1.x;  a[5]=(_Float16)a1.y;  a[6]=(_Float16)a1.z;  a[7]=(_Float16)a1.w;
        a[8]=(_Float16)a2.x;  a[9]=(_Float16)a2.y;  a[10]=(_Float16)a2.z; a[11]=(_Float16)a2.w;
        a[12]=(_Float16)a3.x; a[13]=(_Float16)a3.y; a[14]=(_Float16)a3.z; a[15]=(_Float16)a3.w;
#pragma unroll
        for (int nt = 0; nt < 4; ++nt) {            // 4 output-column tiles (64 cols)
            int col = nt * 16 + m;
            v8h blo = *(const v8h*)&Wt[col * HID + kb];
            v8h bhi = *(const v8h*)&Wt[col * HID + kb + 16];
            v16h b;
#pragma unroll
            for (int j = 0; j < 8; ++j) { b[j] = blo[j]; b[j + 8] = bhi[j]; }
            cacc[nt] = __builtin_amdgcn_wmma_f32_16x16x32_f16(
                false, a, false, b, (short)0, cacc[nt], false, false);
        }
    }
    // C/D layout: VGPR j -> row (j + 8*hs), col = nt*16 + m
#pragma unroll
    for (int nt = 0; nt < 4; ++nt) {
#pragma unroll
        for (int j = 0; j < 8; ++j) {
            T[(size_t)(row0 + hs * 8 + j) * HID + nt * 16 + m] = cacc[nt][j];
        }
    }
}

// ---------- message passing ----------
__global__ __launch_bounds__(256) void k_init_acc(const float* __restrict__ T,
                                                  const float* __restrict__ dinv,
                                                  float* __restrict__ acc, int n) {
    int idx = blockIdx.x * 256 + threadIdx.x;       // node*16 + c4group
    if (idx >= n * 16) return;
    int i = idx >> 4;
    int c4 = (idx & 15) << 2;
    float di = dinv[i];
    float w = di * di;                              // self-loop norm dinv[i]^2
    float4 v = *(const float4*)(T + (size_t)i * HID + c4);
    float4 r; r.x = w * v.x; r.y = w * v.y; r.z = w * v.z; r.w = w * v.w;
    *(float4*)(acc + (size_t)i * HID + c4) = r;
}
__global__ __launch_bounds__(256) void k_edges(const float* __restrict__ T,
                                               float* acc,
                                               const int* __restrict__ src,
                                               const int* __restrict__ dst,
                                               const float* __restrict__ dinv, int e) {
    int tid = blockIdx.x * 256 + threadIdx.x;       // edge*16 + c4group
    int ei = tid >> 4;
    if (ei >= e) return;
    int c4 = (tid & 15) << 2;
    int s = src[ei], d = dst[ei];
    float w = dinv[s] * dinv[d];
    float4 v = *(const float4*)(T + (size_t)s * HID + c4);
    float* ap = acc + (size_t)d * HID + c4;
    atomicAdd(ap + 0, w * v.x);
    atomicAdd(ap + 1, w * v.y);
    atomicAdd(ap + 2, w * v.z);
    atomicAdd(ap + 3, w * v.w);
}
__global__ __launch_bounds__(256) void k_relu_bias(const float* __restrict__ acc,
                                                   const float* __restrict__ b,
                                                   float* __restrict__ Hout, int n) {
    int idx = blockIdx.x * 256 + threadIdx.x;
    if (idx >= n * 16) return;
    int i = idx >> 4;
    int c4 = (idx & 15) << 2;
    float4 v = *(const float4*)(acc + (size_t)i * HID + c4);
    float4 r;
    r.x = fmaxf(v.x + b[c4 + 0], 0.f);
    r.y = fmaxf(v.y + b[c4 + 1], 0.f);
    r.z = fmaxf(v.z + b[c4 + 2], 0.f);
    r.w = fmaxf(v.w + b[c4 + 3], 0.f);
    *(float4*)(Hout + (size_t)i * HID + c4) = r;
}

// ---------- pooling + decoder ----------
__global__ __launch_bounds__(256) void k_zero(float* p, int n) {
    int i = blockIdx.x * 256 + threadIdx.x;
    if (i < n) p[i] = 0.0f;
}
__global__ __launch_bounds__(256) void k_pool(const float* __restrict__ Hc,
                                              const int* __restrict__ batch,
                                              float* Z, int n) {
    int idx = blockIdx.x * 256 + threadIdx.x;       // node*64 + c
    if (idx >= n * HID) return;
    int i = idx >> 6;
    int c = idx & 63;
    int g = batch[i];
    // relu outputs are >= 0 -> IEEE order == unsigned-int order; init 0.0f bits are 0
    atomicMax((unsigned int*)&Z[(size_t)g * HID + c], __float_as_uint(Hc[idx]));
}
__global__ __launch_bounds__(256) void k_decoder(const float* __restrict__ Z,
                                                 const float* __restrict__ W1,
                                                 const float* __restrict__ b1,
                                                 const float* __restrict__ W2,
                                                 const float* __restrict__ b2,
                                                 float* __restrict__ out, int g) {
    int i = blockIdx.x * 256 + threadIdx.x;
    if (i >= g) return;
    const float* z = Z + (size_t)i * HID;
    float y[32];
#pragma unroll
    for (int j = 0; j < 32; ++j) y[j] = b1[j];
    for (int k = 0; k < HID; ++k) {
        float zv = z[k];
#pragma unroll
        for (int j = 0; j < 32; ++j) y[j] = fmaf(zv, W1[k * 32 + j], y[j]);
    }
#pragma unroll
    for (int j = 0; j < 32; ++j) y[j] = fmaxf(y[j], 0.f);
#pragma unroll
    for (int o = 0; o < 4; ++o) {
        float s = b2[o];
#pragma unroll
        for (int j = 0; j < 32; ++j) s = fmaf(y[j], W2[j * 4 + o], s);
        out[(size_t)i * 4 + o] = s;
    }
}

extern "C" void kernel_launch(void* const* d_in, const int* in_sizes, int n_in,
                              void* d_out, int out_size, void* d_ws, size_t ws_size,
                              hipStream_t stream) {
    const float* x     = (const float*)d_in[0];
    const int*   ei    = (const int*)  d_in[1];
    const float* eattr = (const float*)d_in[2];
    const int*   batch = (const int*)  d_in[3];
    const float* Wi    = (const float*)d_in[4];
    const float* bi    = (const float*)d_in[5];
    const float* Wh    = (const float*)d_in[6];
    const float* bh    = (const float*)d_in[7];
    const float* Wd1   = (const float*)d_in[8];
    const float* bd1   = (const float*)d_in[9];
    const float* Wd2   = (const float*)d_in[10];
    const float* bd2   = (const float*)d_in[11];
    (void)n_in; (void)ws_size;

    const int N     = in_sizes[0] / 16;             // 100000
    const int E     = in_sizes[1] / 2;              // 3200000
    const int EA    = in_sizes[2];                  // 12800000 edge_attr floats
    const int G     = (out_size - EA) / 4;          // 1000 graphs
    const int DEPTH = in_sizes[6] / (HID * HID);    // 4
    const int* src = ei;
    const int* dst = ei + E;

    // workspace layout (256B aligned): dinv | T | ACC | H | Z  (~78 MB)
    char* ws = (char*)d_ws;
    size_t off = 0;
    auto bump = [&](size_t bytes) { size_t o = off; off = (off + bytes + 255) & ~(size_t)255; return o; };
    float* dinv = (float*)(ws + bump((size_t)N * 4));
    float* T    = (float*)(ws + bump((size_t)N * HID * 4));
    float* ACC  = (float*)(ws + bump((size_t)N * HID * 4));
    float* Hb   = (float*)(ws + bump((size_t)N * HID * 4));
    float* Z    = (float*)(ws + bump((size_t)G * HID * 4));

    dim3 blk(256);
    const int gN    = (N + 255) / 256;
    const int gE    = (E + 255) / 256;
    const int gNH   = (N * HID + 255) / 256;
    const int gN16  = (N * 16 + 255) / 256;
    const int gE16  = (int)(((size_t)E * 16 + 255) / 256);
    const int gWmma = (N + 127) / 128;              // 8 waves x 16 rows per block
    const int gG    = (G * HID + 255) / 256;

    // normalization
    k_deg_init<<<gN, blk, 0, stream>>>(dinv, N);
    k_deg_count<<<gE, blk, 0, stream>>>(dinv, dst, E);
    k_rsqrt_inplace<<<gN, blk, 0, stream>>>(dinv, N);

    // layer 0 (K=16 input GEMM on VALU, then MP)
    k_gemm_in<<<gNH, blk, 0, stream>>>(x, Wi, T, N);
    k_init_acc<<<gN16, blk, 0, stream>>>(T, dinv, ACC, N);
    k_edges<<<gE16, blk, 0, stream>>>(T, ACC, src, dst, dinv, E);
    k_relu_bias<<<gN16, blk, 0, stream>>>(ACC, bi, Hb, N);

    // hidden layers: WMMA transform + MP
    for (int l = 0; l < DEPTH; ++l) {
        k_gemm_wmma<<<gWmma, blk, 0, stream>>>(Hb, Wh + (size_t)l * HID * HID, T, N);
        k_init_acc<<<gN16, blk, 0, stream>>>(T, dinv, ACC, N);
        k_edges<<<gE16, blk, 0, stream>>>(T, ACC, src, dst, dinv, E);
        k_relu_bias<<<gN16, blk, 0, stream>>>(ACC, bh + (size_t)l * HID, Hb, N);
    }

    // global max pool + decoder MLP
    k_zero<<<gG, blk, 0, stream>>>(Z, G * HID);
    k_pool<<<gNH, blk, 0, stream>>>(Hb, batch, Z, N);
    k_decoder<<<(G + 255) / 256, blk, 0, stream>>>(Z, Wd1, bd1, Wd2, bd2, (float*)d_out, G);

    // second tuple output: edge_attr passthrough
    hipMemcpyAsync((float*)d_out + (size_t)G * 4, eattr, (size_t)EA * sizeof(float),
                   hipMemcpyDeviceToDevice, stream);
}
